// BiParticFusion_57226144252394
// MI455X (gfx1250) — compile-verified, matrix operand order checked by reading, not run on hardware
//
#include <hip/hip_runtime.h>
#include <hip/hip_bf16.h>
#include <math.h>

// ---------------------------------------------------------------------------
// BiParticFusion for MI455X (gfx1250, wave32, WMMA + TDM)
//   v3b: per-block W panel (64 cols x K bf16 <= 64KB) staged into LDS ONCE via
//   the Tensor Data Mover (tensor_load_to_lds + s_wait_tensorcnt), so the
//   WMMA K-loop is pure: 2x global_load_b128 (A) + 8x ds_load_b128 (B) +
//   4x v_wmma_f32_16x16x32_bf16, no barriers, no ds_store.
//   (6-arg tensor_load_to_lds form for the clang-23/therock toolchain.)
// ---------------------------------------------------------------------------

#define TOK   (8 * 4096)   // B*N rows
#define HDIM  256
#define CDIM  512
#define NSEQ  4096
#define NB    8
#define EPSC  1e-6f
#define KMAX  512

typedef __attribute__((ext_vector_type(16))) __bf16 v16bf;
typedef __attribute__((ext_vector_type(8)))  float  v8f;
typedef __attribute__((ext_vector_type(4)))  unsigned int u32x4;
typedef __attribute__((ext_vector_type(8)))  int i32x8;
typedef __attribute__((ext_vector_type(4)))  int i32x4;

__device__ __forceinline__ __bf16 f2bf(float f) {
  union { float f; unsigned u; } in; in.f = f;
  unsigned r = (in.u + 0x7FFFu + ((in.u >> 16) & 1u)) >> 16;   // RNE
  union { unsigned short s; __bf16 b; } out; out.s = (unsigned short)r;
  return out.b;
}

// Tensor Data Mover: DMA a rows x cols (bf16) tile, row stride = stride_elems,
// from global into LDS at lds_off. D# layout per CDNA5 ISA 8.3/8.4.
__device__ __forceinline__ void tdm_load_tile_bf16(const __bf16* gptr, unsigned lds_off,
                                                   int rows, int cols, int stride_elems) {
  unsigned long long ga = (unsigned long long)(uintptr_t)gptr;
  u32x4 g0;
  g0[0] = 1u;                                           // count=1, user mode
  g0[1] = lds_off;                                      // lds_addr
  g0[2] = (unsigned)(ga & 0xFFFFFFFFu);                 // global_addr[31:0]
  g0[3] = (unsigned)((ga >> 32) & 0x01FFFFFFu) | 0x80000000u; // addr[56:32] | type=2
  i32x8 g1;
  g1[0] = (1 << 16);                                    // data_size=1 (2 bytes)
  g1[1] = (cols & 0xFFFF) << 16;                        // tensor_dim0[15:0]
  g1[2] = ((unsigned)cols >> 16) | ((rows & 0xFFFF) << 16); // dim0 hi | tensor_dim1 lo
  g1[3] = ((unsigned)rows >> 16) | ((cols & 0xFFFF) << 16); // dim1 hi | tile_dim0
  g1[4] = rows & 0xFFFF;                                // tile_dim1 (tile_dim2 = 0)
  g1[5] = stride_elems;                                 // tensor_dim0_stride[31:0]
  g1[6] = 0;
  g1[7] = 0;
  i32x4 z4; z4[0] = 0; z4[1] = 0; z4[2] = 0; z4[3] = 0;
  i32x8 z8; z8[0] = 0; z8[1] = 0; z8[2] = 0; z8[3] = 0;
  z8[4] = 0; z8[5] = 0; z8[6] = 0; z8[7] = 0;
  __builtin_amdgcn_tensor_load_to_lds(g0, g1, z4, z4, z8, 0);
}

// ----------------------------- RNG helpers --------------------------------
__device__ __forceinline__ unsigned hash32(unsigned x) {
  x ^= x >> 16; x *= 0x7feb352du;
  x ^= x >> 15; x *= 0x846ca68bu;
  x ^= x >> 16; return x;
}
__device__ __forceinline__ float u01(unsigned h) {
  return (float)(h >> 8) * (1.0f / 16777216.0f) + 1e-7f;
}
__device__ __forceinline__ float2 normal2(unsigned s) {
  unsigned a = hash32(s), b = hash32(s ^ 0x9e3779b9u);
  float u = u01(a), v = u01(b);
  float r = sqrtf(-2.0f * logf(u));
  float sn = sinf(6.28318530718f * v), cs = cosf(6.28318530718f * v);
  return make_float2(r * cs, r * sn);
}

// ---------------------- one-time f32 -> bf16 cast --------------------------
__global__ __launch_bounds__(256)
void cast_bf16_kernel(const float* __restrict__ src, __bf16* __restrict__ dst, int n8) {
  int i = blockIdx.x * 256 + threadIdx.x;
  if (i >= n8) return;
  const float4* s = (const float4*)src + (size_t)i * 2;
  float4 a = s[0], b = s[1];
  union { __bf16 h[8]; uint4 u; } o;
  o.h[0] = f2bf(a.x); o.h[1] = f2bf(a.y); o.h[2] = f2bf(a.z); o.h[3] = f2bf(a.w);
  o.h[4] = f2bf(b.x); o.h[5] = f2bf(b.y); o.h[6] = f2bf(b.z); o.h[7] = f2bf(b.w);
  *(uint4*)(dst + (size_t)i * 8) = o.u;
}

// ---------------------------------------------------------------------------
// GEMM: out[m,n] = epilogue( sum_k A[m,k]*W[n,k] + bias[n] ), all bf16 in.
//   A = concat(A0 [M x K0], A1 [M x K1]); K0/K1 multiples of 32; K <= KMAX.
// Block tile: 128(M) x 64(N), 8 wave32 waves, 16 rows/wave, 4 accum tiles.
// Whole 64 x K W panel TDM'd into LDS once; K-loop has no barriers.
// CDNA5 16-bit A layout: lane<16 holds K={0..7,16..23}, lane>=16 K={8..15,24..31}.
// OP: 0 none | 1 relu | 2 sigmoid | 3 sigmoid*E0 | 4 E0*tanh(.)+(1-E0)*E1
// ---------------------------------------------------------------------------
template <int OP>
__global__ __launch_bounds__(256)
void gemm_bf16_wmma(const __bf16* __restrict__ A0, const __bf16* __restrict__ A1,
                    int K0, int K1,
                    const __bf16* __restrict__ W, const float* __restrict__ bias,
                    const float* __restrict__ E0, const float* __restrict__ E1,
                    float* __restrict__ outF, __bf16* __restrict__ outB, int N) {
  __shared__ __bf16 lwb[64 * KMAX];   // [n 0..63][k 0..K-1] bf16 panel (64 KB max)

  const int tid  = threadIdx.x;
  const int lane = tid & 31;
  const int wave = tid >> 5;
  const int m0   = blockIdx.x * 128 + wave * 16;
  const int n0   = blockIdx.y * 64;
  const int K    = K0 + K1;

  // wave 0 drives the Tensor Data Mover for the whole W panel of this block
  if (tid < 32) {
    tdm_load_tile_bf16(W + (size_t)n0 * K, (unsigned)(uintptr_t)lwb, 64, K, K);
    __builtin_amdgcn_s_wait_tensorcnt(0);
  }
  __syncthreads();

  v8f acc[4];
#pragma unroll
  for (int t = 0; t < 4; ++t)
    acc[t] = (v8f){0.f, 0.f, 0.f, 0.f, 0.f, 0.f, 0.f, 0.f};

  const int hi = lane >> 4;          // 0/1 half-wave k-group select
  const int am = m0 + (lane & 15);   // A row for this lane
  int bbase[4];
#pragma unroll
  for (int t = 0; t < 4; ++t)
    bbase[t] = (t * 16 + (lane & 15)) * K + hi * 8;   // LDS element base per tile

  for (int k = 0; k < K; k += 32) {
    // A fragment: two b128 loads (concat-aware; K0 % 32 == 0)
    const __bf16* arow; int kk;
    if (k < K0) { arow = A0 + (size_t)am * K0; kk = k; }
    else        { arow = A1 + (size_t)am * K1; kk = k - K0; }
    kk += hi * 8;
    union { uint4 u[2]; v16bf v; } af;
    af.u[0] = *(const uint4*)(arow + kk);
    af.u[1] = *(const uint4*)(arow + kk + 16);
    if (k + 32 < K) __builtin_prefetch(arow + kk + 32, 0, 0);

    union { uint4 u[2]; v16bf v; } bf[4];
#pragma unroll
    for (int t = 0; t < 4; ++t) {
      const uint4* bp = (const uint4*)&lwb[bbase[t] + k];
      bf[t].u[0] = bp[0];   // k-group 0..7  (of this half-wave)
      bf[t].u[1] = bp[2];   // k-group 16..23
    }
#pragma unroll
    for (int t = 0; t < 4; ++t)
      acc[t] = __builtin_amdgcn_wmma_f32_16x16x32_bf16(
          false, af.v, false, bf[t].v, (short)0, acc[t], false, false);
  }

  // epilogue: C/D layout — VGPR i: row = m0 + hi*8 + i, col = lane%16
#pragma unroll
  for (int t = 0; t < 4; ++t) {
    const int col = n0 + t * 16 + (lane & 15);
    const float bn = bias[col];
#pragma unroll
    for (int i = 0; i < 8; ++i) {
      const int row = m0 + hi * 8 + i;
      const size_t idx = (size_t)row * N + col;
      float v = acc[t][i] + bn;
      if (OP == 1)      v = fmaxf(v, 0.f);
      else if (OP == 2) v = 1.f / (1.f + __expf(-v));
      else if (OP == 3) v = (1.f / (1.f + __expf(-v))) * E0[idx];
      else if (OP == 4) { float z = E0[idx]; v = z * tanhf(v) + (1.f - z) * E1[idx]; }
      acc[t][i] = v;
    }
  }
  if (outF) {
#pragma unroll
    for (int t = 0; t < 4; ++t) {
      const int col = n0 + t * 16 + (lane & 15);
#pragma unroll
      for (int i = 0; i < 8; ++i)
        outF[(size_t)(m0 + hi * 8 + i) * N + col] = acc[t][i];
    }
  }
  if (outB) {
#pragma unroll
    for (int t = 0; t < 4; ++t) {
      const int col = n0 + t * 16 + (lane & 15);
#pragma unroll
      for (int i = 0; i < 8; ++i)
        outB[(size_t)(m0 + hi * 8 + i) * N + col] = f2bf(acc[t][i]);
    }
  }
}

// ------------------- elementwise / reduction kernels -----------------------
__global__ __launch_bounds__(256)
void wfi_kernel(const float* __restrict__ m1, const float* __restrict__ m2,
                const float* __restrict__ v1, const float* __restrict__ v2,
                float* __restrict__ muw, float* __restrict__ sigw,
                __bf16* __restrict__ muwB, __bf16* __restrict__ sigwB) {
  size_t idx = (size_t)blockIdx.x * 256 + threadIdx.x;
  float a = fmaxf(v1[idx], EPSC), b = fmaxf(v2[idx], EPSC);
  float sw = 1.f / (1.f / a + 1.f / b);
  float c = fmaxf(m1[idx], EPSC), d = fmaxf(m2[idx], EPSC);
  float mw = 1.f / (1.f / c + 1.f / d);
  sigw[idx] = sw; muw[idx] = mw;
  sigwB[idx] = f2bf(sw); muwB[idx] = f2bf(mw);
}

// one block per token: P=2 particle filter + outlier gate, writes fmean, log-var
__global__ __launch_bounds__(256)
void particle_kernel(const float* __restrict__ m1, const float* __restrict__ v1,
                     const float* __restrict__ m2, const float* __restrict__ v2,
                     const float* __restrict__ muw, const float* __restrict__ sigw,
                     const float* __restrict__ nthr,
                     float* __restrict__ fmean, float* __restrict__ fvar) {
  const int tok = blockIdx.x, c = threadIdx.x;
  const size_t idx = (size_t)tok * HDIM + c;
  float M1 = m1[idx], M2 = m2[idx], V1 = v1[idx], V2 = v2[idx];
  float mc = fmaxf(0.5f * (M1 + M2), EPSC);
  float vc = fmaxf(0.5f * (V1 + V2), EPSC);
  float sd = fmaxf(sqrtf(vc + EPSC), EPSC);
  float2 nn = normal2((unsigned)idx * 2654435761u + 12345u);
  float p1 = mc + sd * nn.x, p2 = mc + sd * nn.y;
  float m2e = fmaxf(M2, EPSC), v2e = fmaxf(V2, EPSC);
  float d1 = p1 - m2e, d2 = p2 - m2e;

  __shared__ float red[2][256];
  red[0][c] = d1 * d1 / v2e;
  red[1][c] = d2 * d2 / v2e;
  __syncthreads();
  for (int s = 128; s > 0; s >>= 1) {
    if (c < s) { red[0][c] += red[0][c + s]; red[1][c] += red[1][c + s]; }
    __syncthreads();
  }
  __shared__ float wsh[4];
  if (c == 0) {
    float w1 = fmaxf(__expf(-0.5f * red[0][0]), EPSC) * 0.5f;
    float w2 = fmaxf(__expf(-0.5f * red[1][0]), EPSC) * 0.5f;
    float ws = w1 + w2;
    w1 = fmaxf(w1 / ws, EPSC); w2 = fmaxf(w2 / ws, EPSC);
    float u1 = u01(hash32((unsigned)tok * 747796405u + 1u));
    float u2 = u01(hash32((unsigned)tok * 747796405u + 2u));
    float p0sel = w1 / (w1 + w2);
    wsh[0] = w1; wsh[1] = w2;
    wsh[2] = (u1 < p0sel) ? 0.f : 1.f;
    wsh[3] = (u2 < p0sel) ? 0.f : 1.f;
  }
  __syncthreads();
  float w1 = wsh[0], w2 = wsh[1];
  float pa = (wsh[2] > 0.5f) ? p2 : p1;
  float pb = (wsh[3] > 0.5f) ? p2 : p1;
  float wsum = w1 + w2;
  float fm = (pa * w1 + pb * w2) / wsum;
  float fv = (w1 * (pa - fm) * (pa - fm) + w2 * (pb - fm) * (pb - fm)) / wsum;
  float mw = muw[idx], sw = sigw[idx];
  if (fabsf(fm - mw) > nthr[idx] * sqrtf(sw)) { fm = mw; fv = sw + EPSC; }
  fmean[idx] = fm;                 // aliases m1 (read-before-write per thread)
  fvar[idx]  = logf(fv + EPSC);    // aliases v1
}

// per (b,h): mean over N of fvar and var -> qmeans[b][0..1][h]
__global__ __launch_bounds__(256)
void colmean_kernel(const float* __restrict__ fvar, const float* __restrict__ var,
                    float* __restrict__ qmeans) {
  const int b = blockIdx.x / HDIM, h = blockIdx.x % HDIM, t = threadIdx.x;
  float s0 = 0.f, s1 = 0.f;
  for (int n = t; n < NSEQ; n += 256) {
    size_t idx = ((size_t)b * NSEQ + n) * HDIM + h;
    s0 += fvar[idx]; s1 += var[idx];
  }
  __shared__ float r0[256], r1[256];
  r0[t] = s0; r1[t] = s1; __syncthreads();
  for (int s = 128; s > 0; s >>= 1) {
    if (t < s) { r0[t] += r0[t + s]; r1[t] += r1[t + s]; }
    __syncthreads();
  }
  if (t == 0) {
    qmeans[(size_t)b * 2 * HDIM + h]        = r0[0] * (1.f / NSEQ);
    qmeans[(size_t)b * 2 * HDIM + HDIM + h] = r1[0] * (1.f / NSEQ);
  }
}

// per batch: 2-logit linear + softmax -> wq[b][2]
__global__ __launch_bounds__(256)
void qsoft_kernel(const float* __restrict__ qmeans, const float* __restrict__ qW,
                  const float* __restrict__ qb, float* __restrict__ wq) {
  const int b = blockIdx.x, t = threadIdx.x;
  float s0 = 0.f, s1 = 0.f;
  for (int i = t; i < 2 * HDIM; i += 256) {
    float x = qmeans[(size_t)b * 2 * HDIM + i];
    s0 += x * qW[i]; s1 += x * qW[2 * HDIM + i];
  }
  __shared__ float r0[256], r1[256];
  r0[t] = s0; r1[t] = s1; __syncthreads();
  for (int s = 128; s > 0; s >>= 1) {
    if (t < s) { r0[t] += r0[t + s]; r1[t] += r1[t + s]; }
    __syncthreads();
  }
  if (t == 0) {
    float l0 = r0[0] + qb[0], l1 = r1[0] + qb[1];
    float m = fmaxf(l0, l1);
    float e0 = __expf(l0 - m), e1 = __expf(l1 - m);
    wq[b * 2] = e0 / (e0 + e1); wq[b * 2 + 1] = e1 / (e0 + e1);
  }
}

// fused = eps * exp(0.5*(w0*fvar + w1*var)) + fmean  (bf16 out, feeds final GEMM)
__global__ __launch_bounds__(256)
void reparam_kernel(const float* __restrict__ fvar, const float* __restrict__ var,
                    const float* __restrict__ fmean, const float* __restrict__ wq,
                    __bf16* __restrict__ fusedB) {
  size_t idx = (size_t)blockIdx.x * 256 + threadIdx.x;
  int b = (int)(idx / ((size_t)NSEQ * HDIM));
  float fv = wq[b * 2] * fvar[idx] + wq[b * 2 + 1] * var[idx];
  float2 nn = normal2(0x51ed270bu ^ (unsigned)idx);
  fusedB[idx] = f2bf(nn.x * __expf(0.5f * fv) + fmean[idx]);
}

// ------------------------------ host side ----------------------------------
static inline void launch_gemm(int op, const __bf16* A0, const __bf16* A1, int K0, int K1,
                               const __bf16* W, const float* b, const float* E0,
                               const float* E1, float* outF, __bf16* outB, int N,
                               hipStream_t s) {
  dim3 grid(TOK / 128, N / 64), blk(256);
  switch (op) {
    case 0: gemm_bf16_wmma<0><<<grid, blk, 0, s>>>(A0, A1, K0, K1, W, b, E0, E1, outF, outB, N); break;
    case 1: gemm_bf16_wmma<1><<<grid, blk, 0, s>>>(A0, A1, K0, K1, W, b, E0, E1, outF, outB, N); break;
    case 2: gemm_bf16_wmma<2><<<grid, blk, 0, s>>>(A0, A1, K0, K1, W, b, E0, E1, outF, outB, N); break;
    case 3: gemm_bf16_wmma<3><<<grid, blk, 0, s>>>(A0, A1, K0, K1, W, b, E0, E1, outF, outB, N); break;
    case 4: gemm_bf16_wmma<4><<<grid, blk, 0, s>>>(A0, A1, K0, K1, W, b, E0, E1, outF, outB, N); break;
  }
}

static inline void cast_bf16(const float* src, __bf16* dst, size_t n, hipStream_t s) {
  int n8 = (int)(n / 8);
  cast_bf16_kernel<<<dim3((n8 + 255) / 256), dim3(256), 0, s>>>(src, dst, n8);
}

extern "C" void kernel_launch(void* const* d_in, const int* in_sizes, int n_in,
                              void* d_out, int out_size, void* d_ws, size_t ws_size,
                              hipStream_t stream) {
  const float* f1   = (const float*)d_in[0];
  const float* f2   = (const float*)d_in[1];
  const float* g1pW = (const float*)d_in[2];
  const float* g1pb = (const float*)d_in[3];
  const float* g1gW = (const float*)d_in[4];
  const float* g1gb = (const float*)d_in[5];
  const float* g2pW = (const float*)d_in[6];
  const float* g2pb = (const float*)d_in[7];
  const float* g2gW = (const float*)d_in[8];
  const float* g2gb = (const float*)d_in[9];
  const float* hhW  = (const float*)d_in[10];
  const float* hhb  = (const float*)d_in[11];
  const float* pbW  = (const float*)d_in[12];
  const float* pbb  = (const float*)d_in[13];
  const float* qW   = (const float*)d_in[14];
  const float* qb   = (const float*)d_in[15];
  float* out = (float*)d_out;

  const size_t SLOT = (size_t)TOK * HDIM;       // 8M elements
  float* wsf = (float*)d_ws;
  float* F0 = wsf + 0 * SLOT;                   // a   -> nthr
  float* F1 = wsf + 1 * SLOT;                   // z
  float* F2 = wsf + 2 * SLOT;                   // m1  -> fmean (aliased)
  float* F3 = wsf + 3 * SLOT;                   // v1  -> fvar  (aliased)
  float* F4 = wsf + 4 * SLOT;                   // m2
  float* F5 = wsf + 5 * SLOT;                   // v2
  float* F6 = wsf + 6 * SLOT;                   // muw
  float* F7 = wsf + 7 * SLOT;                   // sigw
  float* F8 = wsf + 8 * SLOT;                   // var
  float* qmeans = wsf + 9 * SLOT;               // B*2H
  float* wq     = qmeans + (size_t)NB * 2 * HDIM;

  __bf16* wsb = (__bf16*)(wq + NB * 2);
  __bf16* f1b = wsb;                    __bf16* f2b = f1b + (size_t)TOK * CDIM;
  __bf16* T0  = f2b + (size_t)TOK * CDIM;       // a_b  -> h1_b -> fused_b
  __bf16* T1  = T0 + SLOT;                      // b2_b -> h2_b
  __bf16* T2  = T1 + SLOT;                      // ra_b -> muw_b
  __bf16* T3  = T2 + SLOT;                      // sigw_b
  __bf16* feat1b = T3 + SLOT;
  __bf16* feat2b = feat1b + SLOT;
  __bf16* g1pWb = feat2b + SLOT;
  __bf16* g1gWb = g1pWb + (size_t)2 * HDIM * CDIM;
  __bf16* g2pWb = g1gWb + (size_t)3 * HDIM * 2 * HDIM;
  __bf16* g2gWb = g2pWb + (size_t)2 * HDIM * CDIM;
  __bf16* hhWb  = g2gWb + (size_t)3 * HDIM * 2 * HDIM;
  __bf16* pbWb  = hhWb + (size_t)8 * HDIM * HDIM;

  const size_t GATE = (size_t)HDIM * 2 * HDIM;
  const size_t HH   = (size_t)HDIM * HDIM;

  // ---- one-time bf16 casts (features + all weights) ----
  cast_bf16(f1,   f1b,   (size_t)TOK * CDIM, stream);
  cast_bf16(f2,   f2b,   (size_t)TOK * CDIM, stream);
  cast_bf16(g1pW, g1pWb, (size_t)2 * HDIM * CDIM, stream);
  cast_bf16(g1gW, g1gWb, (size_t)3 * GATE, stream);
  cast_bf16(g2pW, g2pWb, (size_t)2 * HDIM * CDIM, stream);
  cast_bf16(g2gW, g2gWb, (size_t)3 * GATE, stream);
  cast_bf16(hhW,  hhWb,  (size_t)8 * HH, stream);
  cast_bf16(pbW,  pbWb,  (size_t)CDIM * HDIM, stream);

  // ---- GRU 1: feat1 = gru(f1, f2) ----
  launch_gemm(0, f1b, nullptr, CDIM, 0, g1pWb,             g1pb,        nullptr, nullptr, F0, T0, HDIM, stream); // a
  launch_gemm(0, f2b, nullptr, CDIM, 0, g1pWb + HDIM*CDIM, g1pb + HDIM, nullptr, nullptr, nullptr, T1, HDIM, stream); // b2
  launch_gemm(3, T0, T1, HDIM, HDIM, g1gWb,          g1gb,          F0, nullptr, nullptr, T2, HDIM, stream);     // r*a
  launch_gemm(2, T0, T1, HDIM, HDIM, g1gWb + GATE,   g1gb + HDIM,   nullptr, nullptr, F1, nullptr, HDIM, stream); // z
  launch_gemm(4, T2, T1, HDIM, HDIM, g1gWb + 2*GATE, g1gb + 2*HDIM, F1, F0, nullptr, feat1b, HDIM, stream);      // feat1

  // ---- GRU 2: feat2 = gru(f2, f1) ----
  launch_gemm(0, f2b, nullptr, CDIM, 0, g2pWb,             g2pb,        nullptr, nullptr, F0, T0, HDIM, stream);
  launch_gemm(0, f1b, nullptr, CDIM, 0, g2pWb + HDIM*CDIM, g2pb + HDIM, nullptr, nullptr, nullptr, T1, HDIM, stream);
  launch_gemm(3, T0, T1, HDIM, HDIM, g2gWb,          g2gb,          F0, nullptr, nullptr, T2, HDIM, stream);
  launch_gemm(2, T0, T1, HDIM, HDIM, g2gWb + GATE,   g2gb + HDIM,   nullptr, nullptr, F1, nullptr, HDIM, stream);
  launch_gemm(4, T2, T1, HDIM, HDIM, g2gWb + 2*GATE, g2gb + 2*HDIM, F1, F0, nullptr, feat2b, HDIM, stream);      // feat2

  // ---- hidden heads ----
  launch_gemm(1, feat1b, nullptr, HDIM, 0, hhWb,        hhb,          nullptr, nullptr, nullptr, T0, HDIM, stream); // h1
  launch_gemm(0, T0, nullptr, HDIM, 0, hhWb + 1*HH, hhb + 1*HDIM, nullptr, nullptr, F2, nullptr, HDIM, stream);    // m1
  launch_gemm(0, T0, nullptr, HDIM, 0, hhWb + 2*HH, hhb + 2*HDIM, nullptr, nullptr, F3, nullptr, HDIM, stream);    // v1
  launch_gemm(1, feat2b, nullptr, HDIM, 0, hhWb + 3*HH, hhb + 3*HDIM, nullptr, nullptr, nullptr, T1, HDIM, stream); // h2
  launch_gemm(0, T1, nullptr, HDIM, 0, hhWb + 4*HH, hhb + 4*HDIM, nullptr, nullptr, F4, nullptr, HDIM, stream);    // m2
  launch_gemm(0, T1, nullptr, HDIM, 0, hhWb + 5*HH, hhb + 5*HDIM, nullptr, nullptr, F5, nullptr, HDIM, stream);    // v2

  // ---- weighted fusion init + thresholds ----
  wfi_kernel<<<dim3(TOK * HDIM / 256), dim3(256), 0, stream>>>(F2, F4, F3, F5, F6, F7, T2, T3);
  launch_gemm(0, T2, nullptr, HDIM, 0, hhWb + 6*HH, hhb + 6*HDIM, nullptr, nullptr, F0, nullptr, HDIM, stream); // noise_threshold
  launch_gemm(0, T3, nullptr, HDIM, 0, hhWb + 7*HH, hhb + 7*HDIM, nullptr, nullptr, F8, nullptr, HDIM, stream); // var

  // ---- particle filter (block per token); fmean aliases m1, fvar aliases v1 ----
  particle_kernel<<<dim3(TOK), dim3(256), 0, stream>>>(F2, F3, F4, F5, F6, F7, F0, F2, F3);

  // ---- quality-weighted gate + reparametrize ----
  colmean_kernel<<<dim3(NB * HDIM), dim3(256), 0, stream>>>(F3, F8, qmeans);
  qsoft_kernel<<<dim3(NB), dim3(256), 0, stream>>>(qmeans, qW, qb, wq);
  reparam_kernel<<<dim3(TOK * HDIM / 256), dim3(256), 0, stream>>>(F3, F8, F2, wq, T0); // fused_b

  // ---- final projection H -> C, straight to output ----
  launch_gemm(0, T0, nullptr, HDIM, 0, pbWb, pbb, nullptr, nullptr, out, nullptr, CDIM, stream);
}
